// WeightedMSA_26199300505828
// MI455X (gfx1250) — matrix operand order repaired and checked by reading, not compiled, not run on hardware
//
#include <hip/hip_runtime.h>

// ---------------------------------------------------------------------------
// Weighted multi-head self-attention for MI455X (gfx1250), wave32 + WMMA.
//   qkv = x @ Wqkv^T ; attn = softmax(q k^T / 8) ; out = (softmax(head_w) *
//   (attn @ v)) reshaped @ Wproj^T
// f16 storage, f32 WMMA accumulation (v_wmma_f32_16x16x32_f16).
// Softmax runs in exp2 domain: q is pre-scaled by HDIM^-0.5 * log2(e).
// ---------------------------------------------------------------------------

typedef __attribute__((ext_vector_type(16))) _Float16 v16h;
typedef __attribute__((ext_vector_type(8)))  _Float16 v8h;
typedef __attribute__((ext_vector_type(4)))  _Float16 v4h;
typedef __attribute__((ext_vector_type(8)))  float    v8f;

#define BATCH 2
#define SEQ   2048
#define DIM   1024
#define HEADS 16
#define HDIM  64
#define NQKV  3072                 // 3*DIM
#define QK_SCALE 0.125f            // HDIM^-0.5
#define LOG2E    1.44269504088896f

static __device__ __forceinline__ v16h concat8(v8h lo, v8h hi) {
    return __builtin_shufflevector(lo, hi, 0, 1, 2, 3, 4, 5, 6, 7,
                                           8, 9, 10, 11, 12, 13, 14, 15);
}

// A-operand fragment (16M x 32K, f16). Contiguous K order matching B:
// lane m = lane&15; lanes 0-15 hold K 0..15, lanes 16-31 hold
// K 16..31 (two halfs per VGPR). Source must be K-contiguous rows.
static __device__ __forceinline__ v16h load_frag_a(const _Float16* __restrict__ base,
                                                   int ld, int lmod, int lhalf) {
    const _Float16* p = base + lmod * ld + (lhalf << 4);
    v8h lo = *(const v8h*)(p);        // K +0..7   (relative)
    v8h hi = *(const v8h*)(p + 8);    // K +8..15  (relative)
    return concat8(lo, hi);
}

// B-operand fragment (32K x 16N, f16). Contiguous K order matching A: lane n = lane&15;
// lanes 0-15 hold K 0..15 (2/VGPR), lanes 16-31 hold K 16..31. Source column n
// must be K-contiguous (i.e. row n of the NT operand).
static __device__ __forceinline__ v16h load_frag_b(const _Float16* __restrict__ base,
                                                   int ld, int lmod, int lhalf) {
    const _Float16* p = base + lmod * ld + (lhalf << 4);
    v8h lo = *(const v8h*)(p);
    v8h hi = *(const v8h*)(p + 8);
    return concat8(lo, hi);
}

static __device__ __forceinline__ v8f wmma16(v16h a, v16h b, v8f c) {
    // (neg_a, A, neg_b, B, c_mod, C, reuse_a, reuse_b)
    return __builtin_amdgcn_wmma_f32_16x16x32_f16(false, a, false, b,
                                                  (short)0, c, false, false);
}

static __device__ __forceinline__ v8h pack8(v8f x) {
    v8h r;
#pragma unroll
    for (int i = 0; i < 8; ++i) r[i] = (_Float16)x[i];
    return r;
}

// Build the P B-operand (32K x 16N) from two exp'd S^T C-fragments.
// C layout: lane q holds keys base+(0..7), lane q+16 keys base+(8..15);
// B layout needs lanes 0-15: keys 0..15, lanes 16-31: keys 16..31
// -> swap one 8-half sub-block across the lane halves (4 dword shuffles).
static __device__ __forceinline__ v16h build_pb(v8f s0, v8f s1, int lhalf) {
    union PackU { v8h h; int i[4]; };
    PackU u0, u1, snd, rcv;
    u0.h = pack8(s0);
    u1.h = pack8(s1);
    snd.h = lhalf ? u0.h : u1.h;
#pragma unroll
    for (int t = 0; t < 4; ++t) rcv.i[t] = __shfl_xor(snd.i[t], 16, 32);
    v8h blo = lhalf ? rcv.h : u0.h;
    v8h bhi = lhalf ? u1.h : rcv.h;
    return concat8(blo, bhi);
}

// ---------------------------------------------------------------------------
// f32 -> f16 cast (all sizes are multiples of 1024, vectorize x4)
// ---------------------------------------------------------------------------
__global__ __launch_bounds__(256) void cast_f16_kernel(const float* __restrict__ in,
                                                       _Float16* __restrict__ out,
                                                       int n) {
    int i = (blockIdx.x * 256 + threadIdx.x) * 4;
    if (i + 3 < n) {
        float4 f = *(const float4*)(in + i);
        v4h o = { (_Float16)f.x, (_Float16)f.y, (_Float16)f.z, (_Float16)f.w };
        *(v4h*)(out + i) = o;
    }
}

// ---------------------------------------------------------------------------
// QKV GEMM: qkv[m, d] = sum_c x[m, c] * Wqkv[d, c]   (M=4096, N=3072, K=1024)
// 64x64 tile per wave, 8 waves per block. Scatters into:
//   q[B,H,N,64] (times QK_SCALE*log2e), k[B,H,N,64], vT[B,H,64,N]
// A 64-row tile never crosses the batch boundary (m0 % 64 == 0), so the
// destination base pointers are tile constants; stores use affine immediate
// offsets, and the vT store is contiguous -> one b128 store per fragment.
// ---------------------------------------------------------------------------
__global__ __launch_bounds__(256) void qkv_gemm_kernel(const _Float16* __restrict__ xh,
                                                       const _Float16* __restrict__ wh,
                                                       _Float16* __restrict__ qh,
                                                       _Float16* __restrict__ kh,
                                                       _Float16* __restrict__ vt) {
    const int lane  = threadIdx.x & 31;
    const int wave  = threadIdx.x >> 5;
    const int lmod  = lane & 15;
    const int lhalf = lane >> 4;
    const int tilesN = NQKV / 64;                // 48
    const int gw = blockIdx.x * 8 + wave;
    const int m0 = (gw / tilesN) * 64;
    const int n0 = (gw % tilesN) * 64;

    v8f acc[4][4] = {};
#pragma unroll 1
    for (int k0 = 0; k0 < DIM; k0 += 32) {
        __builtin_prefetch(xh + (size_t)m0 * DIM + k0 + 128, 0, 1);
        __builtin_prefetch(wh + (size_t)n0 * DIM + k0 + 128, 0, 1);
        v16h a[4], b[4];
#pragma unroll
        for (int i = 0; i < 4; ++i)
            a[i] = load_frag_a(xh + (size_t)(m0 + 16 * i) * DIM + k0, DIM, lmod, lhalf);
#pragma unroll
        for (int j = 0; j < 4; ++j)
            b[j] = load_frag_b(wh + (size_t)(n0 + 16 * j) * DIM + k0, DIM, lmod, lhalf);
#pragma unroll
        for (int i = 0; i < 4; ++i)
#pragma unroll
            for (int j = 0; j < 4; ++j)
                acc[i][j] = wmma16(a[i], b[j], acc[i][j]);
    }

    // Scatter. Tile spans exactly one of {q,k,v}, one head, one batch.
    const int part   = n0 / DIM;            // 0=q, 1=k, 2=v
    const int h      = (n0 % DIM) / HDIM;   // head (constant over tile)
    const int bb     = m0 >> 11;            // batch (constant over tile)
    const int nsbase = (m0 & (SEQ - 1)) + 8 * lhalf;

    if (part == 2) {
        // vT[((bb*H + h)*64 + dh)*SEQ + ns], dh = 16j+lmod, ns = nsbase+16i+r
        _Float16* vb = vt + (((size_t)(bb * HEADS + h) * HDIM) + lmod) * SEQ + nsbase;
#pragma unroll
        for (int i = 0; i < 4; ++i)
#pragma unroll
            for (int j = 0; j < 4; ++j)
                *(v8h*)(vb + (size_t)(16 * j) * SEQ + 16 * i) = pack8(acc[i][j]);
    } else {
        _Float16* dst = (part == 0 ? qh : kh)
            + (((size_t)(bb * HEADS + h) * SEQ) + nsbase) * HDIM + lmod;
        const float sc = (part == 0) ? (QK_SCALE * LOG2E) : 1.0f;
#pragma unroll
        for (int i = 0; i < 4; ++i)
#pragma unroll
            for (int j = 0; j < 4; ++j)
#pragma unroll
                for (int r = 0; r < 8; ++r)
                    dst[(size_t)(16 * i + r) * HDIM + 16 * j] =
                        (_Float16)(acc[i][j][r] * sc);
    }
}

// ---------------------------------------------------------------------------
// Transposed flash attention. One wave handles 16 queries of one (b, h):
//   S^T = K . Q^T  (keys = M rows, queries = N cols -> softmax is per-lane)
//   O^T = V^T . P  (A = V^T loads contiguously from the transposed V)
// 64 keys per online-softmax step (4 S^T frags -> 16 WMMAs per step).
// Block = 4 waves = 64 queries. grid = B*H*(SEQ/64) = 1024 blocks.
// ---------------------------------------------------------------------------
__global__ __launch_bounds__(128) void attn_kernel(const _Float16* __restrict__ qh,
                                                   const _Float16* __restrict__ kh,
                                                   const _Float16* __restrict__ vt,
                                                   const float* __restrict__ head_w,
                                                   _Float16* __restrict__ oh) {
    const int lane  = threadIdx.x & 31;
    const int wave  = threadIdx.x >> 5;
    const int lmod  = lane & 15;
    const int lhalf = lane >> 4;

    const int qblocks = SEQ / 64;                    // 32
    const int bb = blockIdx.x / (HEADS * qblocks);
    const int h  = (blockIdx.x / qblocks) % HEADS;
    const int q0 = (blockIdx.x % qblocks) * 64 + wave * 16;

    // softmax(head_w)[h] -- 16 values, computed redundantly (scalar-cheap)
    float hmax = -3.0e38f;
#pragma unroll
    for (int i = 0; i < HEADS; ++i) hmax = fmaxf(hmax, head_w[i]);
    float hsum = 0.f;
#pragma unroll
    for (int i = 0; i < HEADS; ++i) hsum += __expf(head_w[i] - hmax);
    const float hw = __expf(head_w[h] - hmax) / hsum;

    // Q^T B-operand fragments (query = lane column), resident for whole loop
    const _Float16* qbase = qh + ((size_t)(bb * HEADS + h) * SEQ + q0) * HDIM;
    const v16h qb0 = load_frag_b(qbase,      HDIM, lmod, lhalf);  // ch 0..31
    const v16h qb1 = load_frag_b(qbase + 32, HDIM, lmod, lhalf);  // ch 32..63

    const _Float16* kbase0 = kh + (size_t)(bb * HEADS + h) * SEQ * HDIM;
    const _Float16* vbase0 = vt + (size_t)(bb * HEADS + h) * HDIM * SEQ;

    v8f o[4] = {};              // O^T accumulators: 64 dims x 16 queries
    float mrun = -3.0e38f;      // running max (exp2 domain) for query lane&15
    float lrun = 0.0f;          // running denom

#pragma unroll 1
    for (int kt = 0; kt < SEQ; kt += 64) {
        // ---- S^T tile: 64 keys x 16 queries (4 C-frags) --------------------
        v8f st[4];
#pragma unroll
        for (int kg = 0; kg < 4; ++kg) {
            const _Float16* kb = kbase0 + (size_t)(kt + kg * 16) * HDIM;
            v16h ka0 = load_frag_a(kb,      HDIM, lmod, lhalf);
            v16h ka1 = load_frag_a(kb + 32, HDIM, lmod, lhalf);
            v8f s = {};
            s = wmma16(ka0, qb0, s);
            s = wmma16(ka1, qb1, s);
            st[kg] = s;
        }

        // ---- online softmax over keys (per-lane query column, exp2) --------
        float tmax = mrun;
#pragma unroll
        for (int kg = 0; kg < 4; ++kg)
#pragma unroll
            for (int r = 0; r < 8; ++r) tmax = fmaxf(tmax, st[kg][r]);
        tmax = fmaxf(tmax, __shfl_xor(tmax, 16, 32));

        const float corr = __builtin_amdgcn_exp2f(mrun - tmax);
        mrun = tmax;

        float rsum = 0.f;
#pragma unroll
        for (int kg = 0; kg < 4; ++kg)
#pragma unroll
            for (int r = 0; r < 8; ++r) {
                float e = __builtin_amdgcn_exp2f(st[kg][r] - tmax);
                st[kg][r] = e;
                rsum += e;
            }
        rsum += __shfl_xor(rsum, 16, 32);
        lrun = lrun * corr + rsum;

#pragma unroll
        for (int dg = 0; dg < 4; ++dg)
#pragma unroll
            for (int r = 0; r < 8; ++r) o[dg][r] *= corr;

        // ---- P B-operands and O^T += V^T . P -------------------------------
        const v16h pb0 = build_pb(st[0], st[1], lhalf);  // keys kt..kt+31
        const v16h pb1 = build_pb(st[2], st[3], lhalf);  // keys kt+32..kt+63
#pragma unroll
        for (int dg = 0; dg < 4; ++dg) {
            const _Float16* vb = vbase0 + (size_t)(dg * 16) * SEQ + kt;
            v16h va0 = load_frag_a(vb,      SEQ, lmod, lhalf);
            v16h va1 = load_frag_a(vb + 32, SEQ, lmod, lhalf);
            o[dg] = wmma16(va0, pb0, o[dg]);
            o[dg] = wmma16(va1, pb1, o[dg]);
        }
    }

    // ---- normalize, apply head weight, store o[B,N,C] as f16 ---------------
    const float inv = hw / lrun;
#pragma unroll
    for (int dg = 0; dg < 4; ++dg) {
        v8h outv;
#pragma unroll
        for (int r = 0; r < 8; ++r) outv[r] = (_Float16)(o[dg][r] * inv);
        _Float16* dst = oh + ((size_t)(bb * SEQ) + q0 + lmod) * DIM
                           + h * HDIM + dg * 16 + lhalf * 8;
        *(v8h*)dst = outv;   // 8 contiguous dims -> one 16-byte store
    }
}

// ---------------------------------------------------------------------------
// Projection GEMM: out[m, d] = sum_c o[m, c] * Wproj[d, c], f32 output.
// M=4096, N=1024, K=1024. 64x64 tile per wave, 8 waves per block.
// ---------------------------------------------------------------------------
__global__ __launch_bounds__(256) void proj_gemm_kernel(const _Float16* __restrict__ ohh,
                                                        const _Float16* __restrict__ wph,
                                                        float* __restrict__ out) {
    const int lane  = threadIdx.x & 31;
    const int wave  = threadIdx.x >> 5;
    const int lmod  = lane & 15;
    const int lhalf = lane >> 4;
    const int tilesN = DIM / 64;                 // 16
    const int gw = blockIdx.x * 8 + wave;
    const int m0 = (gw / tilesN) * 64;
    const int n0 = (gw % tilesN) * 64;

    v8f acc[4][4] = {};
#pragma unroll 1
    for (int k0 = 0; k0 < DIM; k0 += 32) {
        __builtin_prefetch(ohh + (size_t)m0 * DIM + k0 + 128, 0, 1);
        __builtin_prefetch(wph + (size_t)n0 * DIM + k0 + 128, 0, 1);
        v16h a[4], b[4];
#pragma unroll
        for (int i = 0; i < 4; ++i)
            a[i] = load_frag_a(ohh + (size_t)(m0 + 16 * i) * DIM + k0, DIM, lmod, lhalf);
#pragma unroll
        for (int j = 0; j < 4; ++j)
            b[j] = load_frag_b(wph + (size_t)(n0 + 16 * j) * DIM + k0, DIM, lmod, lhalf);
#pragma unroll
        for (int i = 0; i < 4; ++i)
#pragma unroll
            for (int j = 0; j < 4; ++j)
                acc[i][j] = wmma16(a[i], b[j], acc[i][j]);
    }

    // Hoisted store: base pointer per lane, affine immediate offsets.
    float* ob = out + ((size_t)(m0 + 8 * lhalf)) * DIM + n0 + lmod;
#pragma unroll
    for (int i = 0; i < 4; ++i)
#pragma unroll
        for (int j = 0; j < 4; ++j)
#pragma unroll
            for (int r = 0; r < 8; ++r)
                ob[(size_t)(16 * i + r) * DIM + 16 * j] = acc[i][j][r];
}

// ---------------------------------------------------------------------------
extern "C" void kernel_launch(void* const* d_in, const int* in_sizes, int n_in,
                              void* d_out, int out_size, void* d_ws, size_t ws_size,
                              hipStream_t stream) {
    (void)in_sizes; (void)n_in; (void)out_size; (void)ws_size;
    const float* x      = (const float*)d_in[0];   // [2,2048,1024]
    const float* w_qkv  = (const float*)d_in[1];   // [3072,1024]
    const float* w_proj = (const float*)d_in[2];   // [1024,1024]
    const float* head_w = (const float*)d_in[3];   // [16]
    float* out = (float*)d_out;                    // [2,2048,1024]

    // Workspace layout (halfs): total 24M halfs = 48 MB
    _Float16* xh  = (_Float16*)d_ws;                       // 4096*1024
    _Float16* wqh = xh  + (size_t)4096 * 1024;             // 3072*1024
    _Float16* wph = wqh + (size_t)3072 * 1024;             // 1024*1024
    _Float16* qh  = wph + (size_t)1024 * 1024;             // [B,H,N,64]
    _Float16* kh  = qh  + (size_t)4096 * 1024;             // [B,H,N,64]
    _Float16* vt  = kh  + (size_t)4096 * 1024;             // [B,H,64,N]
    _Float16* oh  = vt  + (size_t)4096 * 1024;             // [B,N,C]

    const int nx = BATCH * SEQ * DIM;      // 4194304
    const int nq = NQKV * DIM;             // 3145728
    const int np = DIM * DIM;              // 1048576

    cast_f16_kernel<<<nx / 1024, 256, 0, stream>>>(x,      xh,  nx);
    cast_f16_kernel<<<nq / 1024, 256, 0, stream>>>(w_qkv,  wqh, nq);
    cast_f16_kernel<<<np / 1024, 256, 0, stream>>>(w_proj, wph, np);

    // 64x48 wave tiles / 8 waves per block = 384 blocks
    qkv_gemm_kernel<<<384, 256, 0, stream>>>(xh, wqh, qh, kh, vt);

    // B*H*(SEQ/64) = 1024 blocks of 4 waves
    attn_kernel<<<1024, 128, 0, stream>>>(qh, kh, vt, head_w, oh);

    // 64x16 wave tiles / 8 = 128 blocks
    proj_gemm_kernel<<<128, 256, 0, stream>>>(oh, wph, out);
}
